// LightHeadRCNNResNet101_Head_5325759447413
// MI455X (gfx1250) — matrix operand order repaired and checked by reading, not compiled, not run on hardware
//
#include <hip/hip_runtime.h>
#include <hip/hip_bf16.h>

// ---------------- geometry ----------------
#define FH   38
#define FW   50
#define PH   52          // FH + 14 halo
#define PW   64          // FW + 14 halo
#define NP   (PH * PW)   // 3328 padded positions
#define NS   (FH * FW)   // 1900 valid positions
#define C_IN 2048
#define C_MID 256
#define C_OUT_P 512      // 490 padded to 512 (32 M-tiles -> 16 M-blocks)
#define C_OUT_REAL 490
#define NROIS 512
#define FC1K 512         // 490 padded
#define FC1N 2048
#define HROWS_P 448      // 336 (loc pad of 324) + 112 (score pad of 81) -> 14 M-blocks
#define LOC_BASE 336

typedef _Float16 h8  __attribute__((ext_vector_type(8)));
typedef _Float16 h16 __attribute__((ext_vector_type(16)));
typedef float    f8  __attribute__((ext_vector_type(8)));

union H16U { h16 v; h8 p[2]; };

// A fragment (16x32 f16, MxK): lane m=l&15, hi=l>>4 -> K = hi*8+{0..7} and 16+hi*8+{0..7}
__device__ __forceinline__ h16 frag_a(const _Float16* __restrict__ p) {
  H16U u; u.p[0] = *(const h8*)p; u.p[1] = *(const h8*)(p + 16); return u.v;
}
// B fragment (32x16 f16, KxN) from [N][K] storage: lane n=l&15, hi -> K = hi*8+{0..7} and 16+hi*8+{0..7}
__device__ __forceinline__ h16 frag_b(const _Float16* __restrict__ p) {
  H16U u; u.p[0] = *(const h8*)p; u.p[1] = *(const h8*)(p + 16); return u.v;
}
#define WMMA_F16(acc, a, b) \
  __builtin_amdgcn_wmma_f32_16x16x32_f16(false, (a), false, (b), (short)0, (acc), false, false)

// ---------------- packing kernels ----------------
// x (1,2048,38,50) f32 -> xP [NP][C_IN] f16 with zero halo
__global__ void pack_xP(const float* __restrict__ x, _Float16* __restrict__ xP) {
  long long idx = (long long)blockIdx.x * blockDim.x + threadIdx.x;
  const long long total = (long long)NP * C_IN;
  if (idx >= total) return;
  int ci = (int)(idx % C_IN);
  int p  = (int)(idx / C_IN);
  int hp = p / PW, wp = p % PW;
  int h = hp - 7, w = wp - 7;
  float v = 0.f;
  if (h >= 0 && h < FH && w >= 0 && w < FW)
    v = x[(size_t)ci * NS + h * FW + w];
  xP[idx] = (_Float16)v;
}

// w_col_max / w_row_max : flat [co][ci][k] f32 -> w1 [2][15][256][2048] f16
__global__ void pack_w1(const float* __restrict__ wc, const float* __restrict__ wr,
                        _Float16* __restrict__ w1) {
  long long idx = (long long)blockIdx.x * blockDim.x + threadIdx.x;
  const long long total = 2LL * 15 * C_MID * C_IN;
  if (idx >= total) return;
  int ci = (int)(idx % C_IN); long long t = idx / C_IN;
  int co = (int)(t % C_MID); t /= C_MID;
  int k  = (int)(t % 15);
  int br = (int)(t / 15);
  const float* src = br ? wr : wc;
  w1[idx] = (_Float16)src[((size_t)co * C_IN + ci) * 15 + k];
}

__global__ void zero_h16(_Float16* __restrict__ dst, long long total) {
  long long idx = (long long)blockIdx.x * blockDim.x + threadIdx.x;
  if (idx < total) dst[idx] = (_Float16)0.f;
}

// w_col / w_row : flat [cf][cm][t] f32 -> w2 [2][15][512][256] f16 (cf padded with 0)
__global__ void pack_w2(const float* __restrict__ wc, const float* __restrict__ wr,
                        _Float16* __restrict__ w2) {
  long long idx = (long long)blockIdx.x * blockDim.x + threadIdx.x;
  const long long total = 2LL * 15 * C_OUT_P * C_MID;
  if (idx >= total) return;
  int cm = (int)(idx % C_MID); long long t = idx / C_MID;
  int cf = (int)(t % C_OUT_P); t /= C_OUT_P;
  int tap = (int)(t % 15);
  int br  = (int)(t / 15);
  const float* src = br ? wr : wc;
  float v = (cf < C_OUT_REAL) ? src[((size_t)cf * C_MID + cm) * 15 + tap] : 0.f;
  w2[idx] = (_Float16)v;
}

// w_fc1 [2048][490] f32 -> [2048][512] f16 (K padded 0)
__global__ void pack_wfc1(const float* __restrict__ w, _Float16* __restrict__ dst) {
  long long idx = (long long)blockIdx.x * blockDim.x + threadIdx.x;
  const long long total = (long long)FC1N * FC1K;
  if (idx >= total) return;
  int k = (int)(idx % FC1K);
  int oc = (int)(idx / FC1K);
  dst[idx] = (_Float16)((k < C_OUT_REAL) ? w[(size_t)oc * C_OUT_REAL + k] : 0.f);
}

// heads: rows 0..335 = w_loc (324 padded), rows 336..447 = w_score (81 padded); K=2048
__global__ void pack_whead(const float* __restrict__ wloc, const float* __restrict__ wscore,
                           _Float16* __restrict__ dst) {
  long long idx = (long long)blockIdx.x * blockDim.x + threadIdx.x;
  const long long total = (long long)HROWS_P * FC1N;
  if (idx >= total) return;
  int k = (int)(idx % FC1N);
  int row = (int)(idx / FC1N);
  float v = 0.f;
  if (row < LOC_BASE) { if (row < 324) v = wloc[(size_t)row * FC1N + k]; }
  else                { int r2 = row - LOC_BASE; if (r2 < 81) v = wscore[(size_t)r2 * FC1N + k]; }
  dst[idx] = (_Float16)v;
}

// ---------------- conv1: 2048 -> 256, 15-tap separable, 2Mx2N register-blocked WMMA ----------------
// out h1P [2][NP][256] f16, bias added
__global__ void conv1_wmma(const _Float16* __restrict__ xP, const _Float16* __restrict__ w1,
                           const float* __restrict__ b_col_max, const float* __restrict__ b_row_max,
                           _Float16* __restrict__ h1P) {
  const int MB = C_MID / 32;             // 8 M-blocks (2 tiles each)
  const int NB = (NS + 31) / 32;         // 60 N-blocks
  int wave = blockIdx.x * (blockDim.x >> 5) + (threadIdx.x >> 5);
  if (wave >= 2 * MB * NB) return;
  int branch = wave / (MB * NB);
  int rem = wave % (MB * NB);
  int mB = rem / NB, nB = rem % NB;
  int lane = threadIdx.x & 31, li = lane & 15, hi = lane >> 4;

  int s0 = nB * 32 + li, s1 = s0 + 16;
  bool v0 = s0 < NS, v1 = s1 < NS;
  int sc0 = v0 ? s0 : NS - 1, sc1 = v1 ? s1 : NS - 1;
  int p0 = (sc0 / FW + 7) * PW + (sc0 % FW + 7);
  int p1 = (sc1 / FW + 7) * PW + (sc1 % FW + 7);
  int mA0 = mB * 32 + li, mA1 = mA0 + 16;

  f8 acc00 = {}, acc01 = {}, acc10 = {}, acc11 = {};
  for (int t = 0; t < 15; ++t) {
    int off = (branch == 0) ? (t - 7) * PW : (t - 7);
    const _Float16* A0 = w1 + ((size_t)((branch * 15 + t) * C_MID + mA0) * C_IN) + hi * 8;
    const _Float16* A1 = w1 + ((size_t)((branch * 15 + t) * C_MID + mA1) * C_IN) + hi * 8;
    const _Float16* B0 = xP + ((size_t)(p0 + off) * C_IN) + hi * 8;
    const _Float16* B1 = xP + ((size_t)(p1 + off) * C_IN) + hi * 8;
    __builtin_prefetch(A0 + C_IN * C_MID, 0, 1);   // next tap -> global_prefetch_b8
    for (int kc = 0; kc < C_IN; kc += 32) {
      h16 a0 = frag_a(A0 + kc), a1 = frag_a(A1 + kc);
      h16 b0 = frag_b(B0 + kc), b1 = frag_b(B1 + kc);
      acc00 = WMMA_F16(acc00, a0, b0);
      acc01 = WMMA_F16(acc01, a0, b1);
      acc10 = WMMA_F16(acc10, a1, b0);
      acc11 = WMMA_F16(acc11, a1, b1);
    }
  }
  const float* bias = (branch == 0) ? b_col_max : b_row_max;
  f8* accs[4] = { &acc00, &acc01, &acc10, &acc11 };
  for (int mi = 0; mi < 2; ++mi) {
    for (int ni = 0; ni < 2; ++ni) {
      f8& a = *accs[mi * 2 + ni];
      h8 out;
      for (int j = 0; j < 8; ++j) {
        int mr = mB * 32 + mi * 16 + hi * 8 + j;
        out[j] = (_Float16)(a[j] + bias[mr]);
      }
      bool vv = ni ? v1 : v0;
      int pp = ni ? p1 : p0;
      if (vv)
        *(h8*)(h1P + ((size_t)(branch * NP + pp) * C_MID) + mB * 32 + mi * 16 + hi * 8) = out;
    }
  }
}

// ---------------- conv2: 256 -> 490(512), both branches fused, + bias + ReLU ----------------
// out hsumP [NP][512] f16
__global__ void conv2_wmma(const _Float16* __restrict__ h1P, const _Float16* __restrict__ w2,
                           const float* __restrict__ b_col, const float* __restrict__ b_row,
                           _Float16* __restrict__ hsumP) {
  const int MB = C_OUT_P / 32;           // 16
  const int NB = (NS + 31) / 32;         // 60
  int wave = blockIdx.x * (blockDim.x >> 5) + (threadIdx.x >> 5);
  if (wave >= MB * NB) return;
  int mB = wave / NB, nB = wave % NB;
  int lane = threadIdx.x & 31, li = lane & 15, hi = lane >> 4;

  int s0 = nB * 32 + li, s1 = s0 + 16;
  bool v0 = s0 < NS, v1 = s1 < NS;
  int sc0 = v0 ? s0 : NS - 1, sc1 = v1 ? s1 : NS - 1;
  int p0 = (sc0 / FW + 7) * PW + (sc0 % FW + 7);
  int p1 = (sc1 / FW + 7) * PW + (sc1 % FW + 7);
  int mA0 = mB * 32 + li, mA1 = mA0 + 16;

  f8 acc00 = {}, acc01 = {}, acc10 = {}, acc11 = {};
  for (int branch = 0; branch < 2; ++branch) {
    for (int t = 0; t < 15; ++t) {
      int off = (branch == 0) ? (t - 7) : (t - 7) * PW;  // col branch: 1x15; row branch: 15x1
      const _Float16* A0 = w2 + ((size_t)((branch * 15 + t) * C_OUT_P + mA0) * C_MID) + hi * 8;
      const _Float16* A1 = w2 + ((size_t)((branch * 15 + t) * C_OUT_P + mA1) * C_MID) + hi * 8;
      const _Float16* B0 = h1P + ((size_t)(branch * NP + p0 + off) * C_MID) + hi * 8;
      const _Float16* B1 = h1P + ((size_t)(branch * NP + p1 + off) * C_MID) + hi * 8;
      __builtin_prefetch(A0 + C_OUT_P * C_MID, 0, 1);
      for (int kc = 0; kc < C_MID; kc += 32) {
        h16 a0 = frag_a(A0 + kc), a1 = frag_a(A1 + kc);
        h16 b0 = frag_b(B0 + kc), b1 = frag_b(B1 + kc);
        acc00 = WMMA_F16(acc00, a0, b0);
        acc01 = WMMA_F16(acc01, a0, b1);
        acc10 = WMMA_F16(acc10, a1, b0);
        acc11 = WMMA_F16(acc11, a1, b1);
      }
    }
  }
  f8* accs[4] = { &acc00, &acc01, &acc10, &acc11 };
  for (int mi = 0; mi < 2; ++mi) {
    for (int ni = 0; ni < 2; ++ni) {
      f8& a = *accs[mi * 2 + ni];
      h8 out;
      for (int j = 0; j < 8; ++j) {
        int mr = mB * 32 + mi * 16 + hi * 8 + j;
        float bias = (mr < C_OUT_REAL) ? (b_col[mr] + b_row[mr]) : 0.f;
        out[j] = (_Float16)fmaxf(a[j] + bias, 0.f);
      }
      bool vv = ni ? v1 : v0;
      int pp = ni ? p1 : p0;
      if (vv)
        *(h8*)(hsumP + (size_t)pp * C_OUT_P + mB * 32 + mi * 16 + hi * 8) = out;
    }
  }
}

// ---------------- PSROI-align style pooling (scalar; tiny FLOPs) ----------------
// pooledP [512 rois][512 (490 padded)] f16
__global__ void roi_pool(const float* __restrict__ rois, const _Float16* __restrict__ hsumP,
                         _Float16* __restrict__ pooledP) {
  int idx = blockIdx.x * blockDim.x + threadIdx.x;
  if (idx >= NROIS * 512) return;
  int r = idx >> 9, k = idx & 511;
  if (k >= C_OUT_REAL) { pooledP[idx] = (_Float16)0.f; return; }
  int b = k / 10;
  int by = b / 7, bx = b % 7;
  const float inv16 = 1.f / 16.f;
  float xmin = rois[r * 4 + 0] * inv16 / FW;
  float ymin = rois[r * 4 + 1] * inv16 / FH;
  float xmax = rois[r * 4 + 2] * inv16 / FW;
  float ymax = rois[r * 4 + 3] * inv16 / FH;
  float step_x = (xmax - xmin) / 7.f;
  float step_y = (ymax - ymin) / 7.f;
  float best = -3.0e38f;
  for (int j = 0; j < 2; ++j) {
    float yy = (ymin + (by + j) * step_y) * (FH - 1);
    float y0f = floorf(yy);
    float fy = yy - y0f;
    int y0 = (int)fminf(fmaxf(y0f, 0.f), (float)(FH - 1));
    int y1 = (int)fminf(fmaxf(y0f + 1.f, 0.f), (float)(FH - 1));
    for (int i = 0; i < 2; ++i) {
      float xx = (xmin + (bx + i) * step_x) * (FW - 1);
      float x0f = floorf(xx);
      float fx = xx - x0f;
      int x0 = (int)fminf(fmaxf(x0f, 0.f), (float)(FW - 1));
      int x1 = (int)fminf(fmaxf(x0f + 1.f, 0.f), (float)(FW - 1));
      float v00 = (float)hsumP[((size_t)((y0 + 7) * PW + x0 + 7)) * C_OUT_P + k];
      float v01 = (float)hsumP[((size_t)((y0 + 7) * PW + x1 + 7)) * C_OUT_P + k];
      float v10 = (float)hsumP[((size_t)((y1 + 7) * PW + x0 + 7)) * C_OUT_P + k];
      float v11 = (float)hsumP[((size_t)((y1 + 7) * PW + x1 + 7)) * C_OUT_P + k];
      float top = v00 + (v01 - v00) * fx;
      float bot = v10 + (v11 - v10) * fx;
      float val = top + (bot - top) * fy;
      best = fmaxf(best, val);
    }
  }
  pooledP[idx] = (_Float16)best;
}

// ---------------- FC1: 490 -> 2048, ReLU, 2Mx2N blocked, out fc1P [512 rois][2048] f16 ----------------
__global__ void fc1_wmma(const _Float16* __restrict__ wA, const _Float16* __restrict__ pooledP,
                         const float* __restrict__ b_fc1, _Float16* __restrict__ fc1P) {
  const int MB = FC1N / 32;    // 64
  const int NB = NROIS / 32;   // 16
  int wave = blockIdx.x * (blockDim.x >> 5) + (threadIdx.x >> 5);
  if (wave >= MB * NB) return;
  int mB = wave / NB, nB = wave % NB;
  int lane = threadIdx.x & 31, li = lane & 15, hi = lane >> 4;
  int mA0 = mB * 32 + li, mA1 = mA0 + 16;
  int r0 = nB * 32 + li, r1 = r0 + 16;

  const _Float16* A0 = wA + (size_t)mA0 * FC1K + hi * 8;
  const _Float16* A1 = wA + (size_t)mA1 * FC1K + hi * 8;
  const _Float16* B0 = pooledP + (size_t)r0 * FC1K + hi * 8;
  const _Float16* B1 = pooledP + (size_t)r1 * FC1K + hi * 8;
  f8 acc00 = {}, acc01 = {}, acc10 = {}, acc11 = {};
  for (int kc = 0; kc < FC1K; kc += 32) {
    h16 a0 = frag_a(A0 + kc), a1 = frag_a(A1 + kc);
    h16 b0 = frag_b(B0 + kc), b1 = frag_b(B1 + kc);
    acc00 = WMMA_F16(acc00, a0, b0);
    acc01 = WMMA_F16(acc01, a0, b1);
    acc10 = WMMA_F16(acc10, a1, b0);
    acc11 = WMMA_F16(acc11, a1, b1);
  }
  f8* accs[4] = { &acc00, &acc01, &acc10, &acc11 };
  for (int mi = 0; mi < 2; ++mi) {
    for (int ni = 0; ni < 2; ++ni) {
      f8& a = *accs[mi * 2 + ni];
      h8 out;
      for (int j = 0; j < 8; ++j) {
        int mr = mB * 32 + mi * 16 + hi * 8 + j;
        out[j] = (_Float16)fmaxf(a[j] + b_fc1[mr], 0.f);
      }
      int rr = ni ? r1 : r0;
      *(h8*)(fc1P + (size_t)rr * FC1N + mB * 32 + mi * 16 + hi * 8) = out;
    }
  }
}

// ---------------- heads: loc (324) + score (81), 2Mx2N blocked, f32 straight to d_out ----------------
__global__ void heads_wmma(const _Float16* __restrict__ wH, const _Float16* __restrict__ fc1P,
                           const float* __restrict__ b_loc, const float* __restrict__ b_score,
                           float* __restrict__ out) {
  const int MB = HROWS_P / 32;  // 14
  const int NB = NROIS / 32;    // 16
  int wave = blockIdx.x * (blockDim.x >> 5) + (threadIdx.x >> 5);
  if (wave >= MB * NB) return;
  int mB = wave / NB, nB = wave % NB;
  int lane = threadIdx.x & 31, li = lane & 15, hi = lane >> 4;
  int mA0 = mB * 32 + li, mA1 = mA0 + 16;
  int r0 = nB * 32 + li, r1 = r0 + 16;

  const _Float16* A0 = wH + (size_t)mA0 * FC1N + hi * 8;
  const _Float16* A1 = wH + (size_t)mA1 * FC1N + hi * 8;
  const _Float16* B0 = fc1P + (size_t)r0 * FC1N + hi * 8;
  const _Float16* B1 = fc1P + (size_t)r1 * FC1N + hi * 8;
  f8 acc00 = {}, acc01 = {}, acc10 = {}, acc11 = {};
  for (int kc = 0; kc < FC1N; kc += 32) {
    h16 a0 = frag_a(A0 + kc), a1 = frag_a(A1 + kc);
    h16 b0 = frag_b(B0 + kc), b1 = frag_b(B1 + kc);
    acc00 = WMMA_F16(acc00, a0, b0);
    acc01 = WMMA_F16(acc01, a0, b1);
    acc10 = WMMA_F16(acc10, a1, b0);
    acc11 = WMMA_F16(acc11, a1, b1);
  }
  f8* accs[4] = { &acc00, &acc01, &acc10, &acc11 };
  for (int mi = 0; mi < 2; ++mi) {
    for (int ni = 0; ni < 2; ++ni) {
      f8& a = *accs[mi * 2 + ni];
      int roi = (ni ? r1 : r0);
      for (int j = 0; j < 8; ++j) {
        int mr = mB * 32 + mi * 16 + hi * 8 + j;
        if (mr < LOC_BASE) {
          if (mr < 324) out[(size_t)roi * 324 + mr] = a[j] + b_loc[mr];
        } else {
          int sc2 = mr - LOC_BASE;
          if (sc2 < 81) out[(size_t)NROIS * 324 + (size_t)roi * 81 + sc2] = a[j] + b_score[sc2];
        }
      }
    }
  }
}

// ---------------- host launcher ----------------
extern "C" void kernel_launch(void* const* d_in, const int* in_sizes, int n_in,
                              void* d_out, int out_size, void* d_ws, size_t ws_size,
                              hipStream_t stream) {
  const float* x         = (const float*)d_in[0];
  const float* rois      = (const float*)d_in[1];
  const float* w_col_max = (const float*)d_in[2];
  const float* b_col_max = (const float*)d_in[3];
  const float* w_col     = (const float*)d_in[4];
  const float* b_col     = (const float*)d_in[5];
  const float* w_row_max = (const float*)d_in[6];
  const float* b_row_max = (const float*)d_in[7];
  const float* w_row     = (const float*)d_in[8];
  const float* b_row     = (const float*)d_in[9];
  const float* w_fc1     = (const float*)d_in[10];
  const float* b_fc1     = (const float*)d_in[11];
  const float* w_score   = (const float*)d_in[12];
  const float* b_score   = (const float*)d_in[13];
  const float* w_loc     = (const float*)d_in[14];
  const float* b_loc     = (const float*)d_in[15];
  float* out = (float*)d_out;

  char* ws = (char*)d_ws;
  // f16 workspace partition (all offsets 256B aligned)
  _Float16* xP    = (_Float16*)(ws + 0);               // 13,631,488 B
  _Float16* w1p   = (_Float16*)(ws + 13631488ULL);     // 31,457,280 B
  _Float16* h1P   = (_Float16*)(ws + 45088768ULL);     //  3,407,872 B
  _Float16* w2p   = (_Float16*)(ws + 48496640ULL);     //  7,864,320 B
  _Float16* hsumP = (_Float16*)(ws + 56360960ULL);     //  3,407,872 B
  _Float16* wf1p  = (_Float16*)(ws + 59768832ULL);     //  2,097,152 B
  _Float16* poolP = (_Float16*)(ws + 61865984ULL);     //    524,288 B
  _Float16* fc1P  = (_Float16*)(ws + 62390272ULL);     //  2,097,152 B
  _Float16* whp   = (_Float16*)(ws + 64487424ULL);     //  1,835,008 B

  auto blk = [](long long n, int t) { return (int)((n + t - 1) / t); };

  // pack / zero
  pack_xP   <<<blk((long long)NP * C_IN, 256), 256, 0, stream>>>(x, xP);
  pack_w1   <<<blk(2LL * 15 * C_MID * C_IN, 256), 256, 0, stream>>>(w_col_max, w_row_max, w1p);
  zero_h16  <<<blk(2LL * NP * C_MID, 256), 256, 0, stream>>>(h1P, 2LL * NP * C_MID);
  pack_w2   <<<blk(2LL * 15 * C_OUT_P * C_MID, 256), 256, 0, stream>>>(w_col, w_row, w2p);
  pack_wfc1 <<<blk((long long)FC1N * FC1K, 256), 256, 0, stream>>>(w_fc1, wf1p);
  pack_whead<<<blk((long long)HROWS_P * FC1N, 256), 256, 0, stream>>>(w_loc, w_score, whp);

  // WMMA pipeline (2Mx2N register-blocked waves, 8 waves / 256-thread block)
  {
    int waves = 2 * (C_MID / 32) * ((NS + 31) / 32);          // 960
    conv1_wmma<<<blk(waves, 8), 256, 0, stream>>>(xP, w1p, b_col_max, b_row_max, h1P);
  }
  {
    int waves = (C_OUT_P / 32) * ((NS + 31) / 32);            // 960
    conv2_wmma<<<blk(waves, 8), 256, 0, stream>>>(h1P, w2p, b_col, b_row, hsumP);
  }
  roi_pool<<<blk((long long)NROIS * 512, 256), 256, 0, stream>>>(rois, hsumP, poolP);
  {
    int waves = (FC1N / 32) * (NROIS / 32);                   // 1024
    fc1_wmma<<<blk(waves, 8), 256, 0, stream>>>(wf1p, poolP, b_fc1, fc1P);
  }
  {
    int waves = (HROWS_P / 32) * (NROIS / 32);                // 224
    heads_wmma<<<blk(waves, 8), 256, 0, stream>>>(whp, fc1P, b_loc, b_score, out);
  }
}
